// ContrastiveMoCoKnnBert_53369263620108
// MI455X (gfx1250) — matrix operand order repaired and checked by reading, not compiled, not run on hardware
//
#include <hip/hip_runtime.h>
#include <math.h>

// ---------------------------------------------------------------------------
// MI455X (gfx1250) implementation.
//
// Roofline: feature_queue = 192 MB fp32 streamed once -> 8.3 us HBM floor at
// 23.3 TB/s. GEMM is 6.4 GFLOP (33 FLOP/byte) so fp32 math would need
// ~770 TFLOPS to keep up; bf16 WMMA 16x16x32 (f32 accumulate) keeps compute
// far below the HBM floor, so everything runs through v_wmma_f32_16x16x32_bf16.
// Single pass over fq fused with per-row top-25 + masked sum(exp); all
// reductions are fixed-order (no float atomics) => deterministic.
//
// Workspace requirement: ~4.2 MB (see offsets in kernel_launch).
// ---------------------------------------------------------------------------

typedef __attribute__((ext_vector_type(16))) __bf16 v16bf;
typedef __attribute__((ext_vector_type(8)))  float  v8f;

#define HID   768
#define QK    65536
#define NBLK  512          // QK / 128 columns per block
#define TOPK  25
#define NLOG  63           // NUM_LABELS - 1
// 1/T = 2.0f, RATE = 0.5

__device__ __forceinline__ v8f vzero8() {
  v8f z;
#pragma unroll
  for (int i = 0; i < 8; ++i) z[i] = 0.0f;
  return z;
}

// ---------------------------------------------------------------------------
// out[64 x Nact] = act(A[64x768] @ W[768xNact] + bias),  act: 0=none 1=tanh
// 128 threads (4 waves); wave w owns output cols [blk*64 + 16w, +16),
// 4 M-tiles of 16 rows each. bf16 fragments built on the fly from f32.
//
// A-fragment layout (16-bit A 16x32, ISA 7.12.2): lane l (m=l&15, half=l>>4):
//   vec[0..7]  = A[m][k0 + 8*half     .. +7]
//   vec[8..15] = A[m][k0 + 16 + 8*half.. +7]
// B-fragment (32x16): lane l (n=l&15, half=l>>4): vec[0..7] = B[k0+8*half+e][n], vec[8..15] = B[k0+16+8*half+e-8][n]
// C/D f32 16x16: lane l: c[v] = D[v + 8*(l>>4)][l&15]
// ---------------------------------------------------------------------------
__global__ void gemm64_bias_act(const float* __restrict__ A,
                                const float* __restrict__ W,
                                const float* __restrict__ bias,
                                float* __restrict__ out,
                                int Nact, int act) {
  const int lane = threadIdx.x & 31;
  const int wave = threadIdx.x >> 5;
  const int half = lane >> 4;
  const int l15  = lane & 15;
  const int n    = blockIdx.x * 64 + wave * 16 + l15;
  const bool nok = (n < Nact);

  v8f acc[4];
#pragma unroll
  for (int t = 0; t < 4; ++t) acc[t] = vzero8();

  for (int kk = 0; kk < HID; kk += 32) {
    v16bf bfrag;
#pragma unroll
    for (int e = 0; e < 16; ++e) {
      const int k = kk + ((e < 8) ? (8 * half + e) : (8 + 8 * half + e));
      const float v = nok ? W[(size_t)k * (size_t)Nact + n] : 0.0f;
      bfrag[e] = (__bf16)v;
    }
#pragma unroll
    for (int t = 0; t < 4; ++t) {
      const float* ar = A + (size_t)(t * 16 + l15) * HID;
      v16bf afrag;
#pragma unroll
      for (int e = 0; e < 8; ++e) afrag[e]     = (__bf16)ar[kk + 8 * half + e];
#pragma unroll
      for (int e = 0; e < 8; ++e) afrag[8 + e] = (__bf16)ar[kk + 16 + 8 * half + e];
      acc[t] = __builtin_amdgcn_wmma_f32_16x16x32_bf16(
          false, afrag, false, bfrag, (short)0, acc[t], false, false);
    }
  }

  const float bv = nok ? bias[n] : 0.0f;
#pragma unroll
  for (int t = 0; t < 4; ++t) {
#pragma unroll
    for (int v = 0; v < 8; ++v) {
      const int m = t * 16 + v + 8 * half;
      float x = acc[t][v] + bv;
      if (act == 1) x = tanhf(x);
      if (nok) out[(size_t)m * (size_t)Nact + n] = x;
    }
  }
}

// ---------------------------------------------------------------------------
// liner_q = LP / ||LP_row||, stored as bf16 for the big WMMA GEMM.
// ---------------------------------------------------------------------------
__global__ void rownorm64(const float* __restrict__ LP, __bf16* __restrict__ LQ) {
  const int r = blockIdx.x;
  __shared__ float red[256];
  float s = 0.0f;
  for (int c = threadIdx.x; c < HID; c += 256) {
    const float v = LP[r * HID + c];
    s += v * v;
  }
  red[threadIdx.x] = s;
  __syncthreads();
  for (int off = 128; off > 0; off >>= 1) {
    if (threadIdx.x < off) red[threadIdx.x] += red[threadIdx.x + off];
    __syncthreads();
  }
  const float inv = 1.0f / sqrtf(red[0]);
  for (int c = threadIdx.x; c < HID; c += 256)
    LQ[r * HID + c] = (__bf16)(LP[r * HID + c] * inv);
}

// ---------------------------------------------------------------------------
// Big fused pass: cos tile [64 x 128] via WMMA, then per-row top-25 candidates
// and masked sum(exp(2*cos)) for this block's columns.
// B = fq^T, so each lane's B fragment is 16 CONSECUTIVE f32 of one fq row.
// ---------------------------------------------------------------------------
__global__ void cos_topk(const __bf16* __restrict__ LQ,      // [64 x 768] bf16
                         const float*  __restrict__ FQ,      // [65536 x 768] f32
                         const int*    __restrict__ labels,  // [64]
                         const long long* __restrict__ lqueue,// [65536]
                         float* __restrict__ cand,           // [64][NBLK*TOPK]
                         float* __restrict__ sneg)           // [64][NBLK]
{
  const int blk  = blockIdx.x;          // 512 blocks x 128 queue columns
  const int lane = threadIdx.x & 31;
  const int wave = threadIdx.x >> 5;    // 8 waves
  const int half = lane >> 4;
  const int l15  = lane & 15;
  const int n    = blk * 128 + wave * 16 + l15;   // fq row index

  __shared__ float tile[64][129];       // pitch 129 -> conflict-free col reads
  __shared__ float tk[64][TOPK];
  __shared__ int   lqs[128];

  v8f acc[4];
#pragma unroll
  for (int t = 0; t < 4; ++t) acc[t] = vzero8();

  const float* fr = FQ + (size_t)n * HID;
  for (int kk = 0; kk < HID; kk += 32) {
    if (kk + 32 < HID) __builtin_prefetch(fr + kk + 32 + 16 * half, 0, 1);
    const float4* p = (const float4*)(fr + kk + 16 * half);
    const float4 x0 = p[0], x1 = p[1], x2 = p[2], x3 = p[3];
    v16bf bfrag;
    bfrag[0]  = (__bf16)x0.x; bfrag[1]  = (__bf16)x0.y;
    bfrag[2]  = (__bf16)x0.z; bfrag[3]  = (__bf16)x0.w;
    bfrag[4]  = (__bf16)x1.x; bfrag[5]  = (__bf16)x1.y;
    bfrag[6]  = (__bf16)x1.z; bfrag[7]  = (__bf16)x1.w;
    bfrag[8]  = (__bf16)x2.x; bfrag[9]  = (__bf16)x2.y;
    bfrag[10] = (__bf16)x2.z; bfrag[11] = (__bf16)x2.w;
    bfrag[12] = (__bf16)x3.x; bfrag[13] = (__bf16)x3.y;
    bfrag[14] = (__bf16)x3.z; bfrag[15] = (__bf16)x3.w;
#pragma unroll
    for (int t = 0; t < 4; ++t) {
      const __bf16* ar = LQ + (size_t)(t * 16 + l15) * HID;
      v16bf afrag;
#pragma unroll
      for (int e = 0; e < 8; ++e) afrag[e]     = ar[kk + 16 * half + e];
#pragma unroll
      for (int e = 0; e < 8; ++e) afrag[8 + e] = ar[kk + 16 * half + 8 + e];
      acc[t] = __builtin_amdgcn_wmma_f32_16x16x32_bf16(
          false, afrag, false, bfrag, (short)0, acc[t], false, false);
    }
  }

  // Spill 64x128 cos tile to LDS.
#pragma unroll
  for (int t = 0; t < 4; ++t)
#pragma unroll
    for (int v = 0; v < 8; ++v)
      tile[t * 16 + v + 8 * half][wave * 16 + l15] = acc[t][v];

  if (threadIdx.x < 128)
    lqs[threadIdx.x] = (int)lqueue[blk * 128 + threadIdx.x];
  __syncthreads();

  // Row-owner threads: top-25 candidates (raw cos) + masked sum of exp(cos/T).
  if (threadIdx.x < 64) {
    const int r   = threadIdx.x;
    const int lab = labels[r];
#pragma unroll
    for (int j = 0; j < TOPK; ++j) tk[r][j] = -INFINITY;
    float s = 0.0f;
    for (int c = 0; c < 128; ++c) {
      const float v = tile[r][c];
      if (lqs[c] != lab) s += expf(2.0f * v);     // 1/T = 2
      if (v > tk[r][TOPK - 1]) {
        int j = TOPK - 1;
        while (j > 0 && tk[r][j - 1] < v) { tk[r][j] = tk[r][j - 1]; --j; }
        tk[r][j] = v;
      }
    }
#pragma unroll
    for (int j = 0; j < TOPK; ++j)
      cand[(size_t)r * (NBLK * TOPK) + blk * TOPK + j] = tk[r][j];
    sneg[r * NBLK + blk] = s;
  }
}

// ---------------------------------------------------------------------------
// Final deterministic reduce: merge candidates -> global top-25 per row,
// sum neg partials, compute loss_con + loss_cls, combine.
// ---------------------------------------------------------------------------
__global__ void finalize64(const float* __restrict__ cand,
                           const float* __restrict__ sneg,
                           const float* __restrict__ LC,     // [64 x 63]
                           const int*   __restrict__ labels,
                           float* __restrict__ out) {
  const int b = threadIdx.x;  // 64 threads
  __shared__ float tk[64][TOPK];
  __shared__ float conr[64], cer[64];

#pragma unroll
  for (int j = 0; j < TOPK; ++j) tk[b][j] = -INFINITY;
  const float* cr = cand + (size_t)b * (NBLK * TOPK);
  for (int i = 0; i < NBLK * TOPK; ++i) {
    const float v = cr[i];
    if (v > tk[b][TOPK - 1]) {
      int j = TOPK - 1;
      while (j > 0 && tk[b][j - 1] < v) { tk[b][j] = tk[b][j - 1]; --j; }
      tk[b][j] = v;
    }
  }

  double sn = 0.0;
  for (int i = 0; i < NBLK; ++i) sn += (double)sneg[b * NBLK + i];
  const float Sneg = (float)sn;

  // loss_con row contribution: sum_j [ log(exp(p/T) + Sneg) - p/T ]
  float con = 0.0f;
  for (int j = 0; j < TOPK; ++j) {
    const float p2 = 2.0f * tk[b][j];
    con += logf(expf(p2) + Sneg) - p2;
  }

  // CE over 63 classes.
  const float* lr = LC + b * NLOG;
  float m = -INFINITY;
  for (int c = 0; c < NLOG; ++c) m = fmaxf(m, lr[c]);
  float se = 0.0f;
  for (int c = 0; c < NLOG; ++c) se += expf(lr[c] - m);
  const float ce = (m + logf(se)) - lr[labels[b]];

  conr[b] = con;
  cer[b]  = ce;
  __syncthreads();
  if (b == 0) {
    float sc = 0.0f, sl = 0.0f;
    for (int i = 0; i < 64; ++i) { sc += conr[i]; sl += cer[i]; }
    const float loss_con = sc / (64.0f * (float)TOPK);
    const float loss_cls = sl / 64.0f;
    out[0] = 0.5f * loss_con + 0.5f * loss_cls;
  }
}

// ---------------------------------------------------------------------------
extern "C" void kernel_launch(void* const* d_in, const int* in_sizes, int n_in,
                              void* d_out, int out_size, void* d_ws, size_t ws_size,
                              hipStream_t stream) {
  const float*     q   = (const float*)d_in[0];
  const int*       lab = (const int*)d_in[1];
  const long long* lq  = (const long long*)d_in[2];
  const float*     fq  = (const float*)d_in[3];
  const float*     Wd  = (const float*)d_in[4];
  const float*     bd  = (const float*)d_in[5];
  const float*     Wo  = (const float*)d_in[6];
  const float*     bo  = (const float*)d_in[7];
  const float*     Wc1 = (const float*)d_in[8];
  const float*     bc1 = (const float*)d_in[9];
  const float*     Wc2 = (const float*)d_in[10];
  const float*     bc2 = (const float*)d_in[11];
  float*           out = (float*)d_out;

  // Workspace layout (bytes). Total needed: 4,112,128 B (~4.0 MiB).
  char*   ws = (char*)d_ws;
  float*  H1 = (float*)(ws + 0);        // 64x768 f32   tanh(q@Wd+bd)
  float*  H2 = (float*)(ws + 196608);   // 64x768 f32   tanh(q@Wc1+bc1)
  float*  LP = (float*)(ws + 393216);   // 64x768 f32   H1@Wo+bo (pre-norm)
  float*  LC = (float*)(ws + 589824);   // 64x63  f32   logits_cls
  __bf16* LQ = (__bf16*)(ws + 605952);  // 64x768 bf16  l2norm(LP)
  float*  SN = (float*)(ws + 704256);   // 64x512 f32   per-block neg expsum
  float*  CD = (float*)(ws + 835328);   // 64x12800 f32 per-block top-25 cand

  gemm64_bias_act<<<12, 128, 0, stream>>>(q,  Wd,  bd,  H1, HID,  1);
  gemm64_bias_act<<<12, 128, 0, stream>>>(q,  Wc1, bc1, H2, HID,  1);
  gemm64_bias_act<<<12, 128, 0, stream>>>(H1, Wo,  bo,  LP, HID,  0);
  gemm64_bias_act<<<1,  128, 0, stream>>>(H2, Wc2, bc2, LC, NLOG, 0);
  rownorm64<<<64, 256, 0, stream>>>(LP, LQ);
  cos_topk<<<NBLK, 256, 0, stream>>>(LQ, fq, lab, lq, CD, SN);
  finalize64<<<1, 64, 0, stream>>>(CD, SN, LC, lab, out);
}
